// linear_GRU_6399501271097
// MI455X (gfx1250) — matrix operand order repaired and checked by reading, not compiled, not run on hardware
//
#include <hip/hip_runtime.h>
#include <hip/hip_fp16.h>

typedef __attribute__((ext_vector_type(16))) _Float16 v16h;
typedef __attribute__((ext_vector_type(8)))  _Float16 v8h;
typedef __attribute__((ext_vector_type(4)))  _Float16 v4h;
typedef __attribute__((ext_vector_type(8)))  float    v8f;

// ---------------------------------------------------------------------------
// Transpose + convert f32 [K,N] row-major -> f16 [N,K] row-major (LDS tiled).
// K and N must be multiples of 32.  block = (32,8)
// ---------------------------------------------------------------------------
__global__ void transpose_to_f16(const float* __restrict__ in,
                                 _Float16* __restrict__ out, int K, int N) {
  __shared__ float tile[32][33];
  const int k0 = blockIdx.y * 32;
  const int n0 = blockIdx.x * 32;
  const int tx = threadIdx.x;   // 0..31
  const int ty = threadIdx.y;   // 0..7
#pragma unroll
  for (int i = 0; i < 32; i += 8) {
    tile[ty + i][tx] = in[(size_t)(k0 + ty + i) * N + (n0 + tx)];
  }
  __syncthreads();
#pragma unroll
  for (int i = 0; i < 32; i += 8) {
    out[(size_t)(n0 + ty + i) * K + (k0 + tx)] = (_Float16)tile[tx][ty + i];
  }
}

// nb[b,d] = (f16) b_out[d]   (grid covers B*D elements)
__global__ void init_nb(const float* __restrict__ b_out,
                        _Float16* __restrict__ nb, int D) {
  int idx = blockIdx.x * blockDim.x + threadIdx.x;
  nb[idx] = (_Float16)b_out[idx % D];
}

// diff[b,d] = (f16)(x[b,t,d] - nb[b,d]) , 4 elements per thread
__global__ void diff_kernel(const float* __restrict__ x,
                            const _Float16* __restrict__ nb,
                            _Float16* __restrict__ diff,
                            int t, int T, int D) {
  int idx = (blockIdx.x * blockDim.x + threadIdx.x) * 4;  // over B*D
  int b = idx / D;
  int d = idx - b * D;
  const float4 xv = *(const float4*)(x + ((size_t)b * T + t) * D + d);
  v4h nv = *(const v4h*)(nb + idx);
  v4h dv;
  dv[0] = (_Float16)(xv.x - (float)nv[0]);
  dv[1] = (_Float16)(xv.y - (float)nv[1]);
  dv[2] = (_Float16)(xv.z - (float)nv[2]);
  dv[3] = (_Float16)(xv.w - (float)nv[3]);
  *(v4h*)(diff + idx) = dv;
}

// ---------------------------------------------------------------------------
// WMMA GEMM:  C[M,N] = A[M,K] @ Wt[N,K]^T  (+ bias), f16 in / f32 acc.
// Each wave computes a 32x32 output tile: 2 m-tiles x 2 n-tiles, so both
// A-fragments are reused across both B-fragments (4 WMMAs per 8 b128 loads).
// block = 256 threads = 8 waves; wave w handles n-block (blockIdx.y*8 + w)*32.
// MODE 0: store f16 hidden.  MODE 1: sigmoid -> f32 out[:,t,:] + f16 nb.
// ---------------------------------------------------------------------------
template <int MODE>
__global__ __launch_bounds__(256) void
gemm_wmma(const _Float16* __restrict__ A,    // [M,K] f16
          const _Float16* __restrict__ Wt,   // [N,K] f16 (transposed weight)
          const float*    __restrict__ bias, // [N]
          _Float16*       __restrict__ outH, // [M,N] f16
          float*          __restrict__ outF, // [M,T,N] f32 (MODE 1 only)
          int N, int K, int T, int t) {
  constexpr int MT = 2;                        // 16-row tiles per wave
  constexpr int NT = 2;                        // 16-col tiles per wave
  const int lane = threadIdx.x & 31;
  const int wave = threadIdx.x >> 5;
  const int m0   = blockIdx.x * (16 * MT);
  const int n0   = (blockIdx.y * 8 + wave) * (16 * NT);

  const int lrow  = lane & 15;                 // matrix row/col within tile
  const int akb   = (lane < 16) ? 0 : 16;      // A-frag K base (contiguous k = 16h + s)
  const int bkb   = (lane < 16) ? 0 : 16;      // B-frag K base (contiguous k = 16h + s)

  const _Float16* bptr0 = Wt + (size_t)(n0 + lrow) * K + bkb;
  const _Float16* bptr1 = Wt + (size_t)(n0 + 16 + lrow) * K + bkb;
  const _Float16* aptr0 = A + (size_t)(m0 + lrow) * K + akb;
  const _Float16* aptr1 = A + (size_t)(m0 + 16 + lrow) * K + akb;

  v8f acc[MT][NT];
#pragma unroll
  for (int i = 0; i < MT; ++i)
#pragma unroll
    for (int j = 0; j < NT; ++j) acc[i][j] = (v8f){};

  for (int kk = 0; kk < K; kk += 32) {
    v16h bf[NT], af[MT];
    {
      v8h blo0 = *(const v8h*)(bptr0 + kk);
      v8h bhi0 = *(const v8h*)(bptr0 + kk + 8);
      v8h blo1 = *(const v8h*)(bptr1 + kk);
      v8h bhi1 = *(const v8h*)(bptr1 + kk + 8);
      v8h alo0 = *(const v8h*)(aptr0 + kk);
      v8h ahi0 = *(const v8h*)(aptr0 + kk + 8);
      v8h alo1 = *(const v8h*)(aptr1 + kk);
      v8h ahi1 = *(const v8h*)(aptr1 + kk + 8);
#pragma unroll
      for (int e = 0; e < 8; ++e) {
        bf[0][e] = blo0[e]; bf[0][e + 8] = bhi0[e];
        bf[1][e] = blo1[e]; bf[1][e + 8] = bhi1[e];
        af[0][e] = alo0[e]; af[0][e + 8] = ahi0[e];
        af[1][e] = alo1[e]; af[1][e + 8] = ahi1[e];
      }
    }
#pragma unroll
    for (int i = 0; i < MT; ++i)
#pragma unroll
      for (int j = 0; j < NT; ++j)
        acc[i][j] = __builtin_amdgcn_wmma_f32_16x16x32_f16(
            false, af[i], false, bf[j], (short)0, acc[i][j], false, false);
  }

  const int mhalf = (lane >> 4) * 8;           // C/D layout: lanes 16-31 -> M+8
#pragma unroll
  for (int j = 0; j < NT; ++j) {
    const int ncol = n0 + 16 * j + lrow;
    const float bv = bias[ncol];
#pragma unroll
    for (int i = 0; i < MT; ++i) {
#pragma unroll
      for (int r = 0; r < 8; ++r) {
        const int m = m0 + 16 * i + mhalf + r;
        float v = acc[i][j][r] + bv;
        if (MODE == 0) {
          outH[(size_t)m * N + ncol] = (_Float16)v;
        } else {
          float p = 1.0f / (1.0f + __expf(-v));
          outF[((size_t)m * T + t) * N + ncol] = p;   // preds[b, t, :]
          outH[(size_t)m * N + ncol] = (_Float16)p;   // carry for next step
        }
      }
    }
  }
}

// ---------------------------------------------------------------------------
extern "C" void kernel_launch(void* const* d_in, const int* in_sizes, int n_in,
                              void* d_out, int out_size, void* d_ws, size_t ws_size,
                              hipStream_t stream) {
  const float* x        = (const float*)d_in[0];  // [B,T,D]
  const float* Z        = (const float*)d_in[1];  // [H,D]
  const float* b_out_p  = (const float*)d_in[2];  // [D]
  const float* KFinv    = (const float*)d_in[3];  // [D,H]
  const float* b_hidden = (const float*)d_in[4];  // [H]
  float* out = (float*)d_out;

  const int D = in_sizes[2];                 // 4096
  const int H = in_sizes[4];                 // 1024
  const int T = 100;
  const int B = in_sizes[0] / (T * D);       // 256

  // Workspace layout (all f16): KT[H,D] | ZT[D,H] | nb[B,D] | diff[B,D] | hid[B,H]
  char* ws = (char*)d_ws;
  _Float16* KT   = (_Float16*)ws;                                   // 8 MB
  _Float16* ZT   = (_Float16*)(ws + (size_t)H * D * 2);             // 8 MB
  _Float16* nb   = (_Float16*)(ws + (size_t)H * D * 4);             // 2 MB
  _Float16* diff = nb + (size_t)B * D;                              // 2 MB
  _Float16* hid  = diff + (size_t)B * D;                            // 0.5 MB

  // One-time (per launch) weight conversion + state init
  transpose_to_f16<<<dim3(H / 32, D / 32), dim3(32, 8), 0, stream>>>(KFinv, KT, D, H);
  transpose_to_f16<<<dim3(D / 32, H / 32), dim3(32, 8), 0, stream>>>(Z, ZT, H, D);
  init_nb<<<(B * D) / 256, 256, 0, stream>>>(b_out_p, nb, D);

  for (int t = 0; t < T; ++t) {
    diff_kernel<<<(B * D / 4) / 256, 256, 0, stream>>>(x, nb, diff, t, T, D);
    // hidden = diff @ KFinv + b_hidden        (M=B, N=H, K=D)
    gemm_wmma<0><<<dim3(B / 32, H / 256), 256, 0, stream>>>(
        diff, KT, b_hidden, hid, nullptr, H, D, T, t);
    // nb = sigmoid(hidden @ Z + b_out)        (M=B, N=D, K=H)
    gemm_wmma<1><<<dim3(B / 32, D / 256), 256, 0, stream>>>(
        hid, ZT, b_out_p, nb, out, D, H, T, t);
  }
}
